// MultiheadSelfAttention_84585085927772
// MI455X (gfx1250) — matrix operand order repaired and checked by reading, not compiled, not run on hardware
//
#include <hip/hip_runtime.h>
#include <cstdint>
#include <cstddef>

// ---------------------------------------------------------------------------
// MI455X (gfx1250) multi-head causal self-attention with fractional RoPE.
// GEMMs via V_WMMA_F32_16X16X32_BF16 (wave32, f32 accumulate); attention K/V
// tiles staged into LDS by the Tensor Data Mover (tensor_load_to_lds,
// TENSORcnt), double-buffered, with pad_enable for bank-conflict-free reads.
// ---------------------------------------------------------------------------

typedef __attribute__((ext_vector_type(16))) __bf16 v16bf;
typedef __attribute__((ext_vector_type(8)))  __bf16 v8bf;
typedef __attribute__((ext_vector_type(8)))  float  v8f;
typedef __attribute__((ext_vector_type(4)))  unsigned int v4u;
typedef __attribute__((ext_vector_type(8)))  int v8i_;
typedef __attribute__((ext_vector_type(4)))  int v4i_;

constexpr int Bc = 2;
constexpr int Tc = 2048;
constexpr int Mc = 1024;     // d_model
constexpr int Hc = 8;
constexpr int Dc = 128;      // head dim
constexpr float kScale  = 1.0f / 128.0f;   // mu-param: 1/HEAD_DIM
constexpr float kNegInf = -1e30f;

// Padded LDS row strides (elements) chosen for conflict-free ds_load_b128:
// K: 128+8 = 136 elems = 272B (bank stride 4);  V: 32+8 = 40 elems = 80B.
constexpr int kKRow = 136;
constexpr int kVRow = 40;

#if defined(__HIP_DEVICE_COMPILE__) && \
    __has_builtin(__builtin_amdgcn_tensor_load_to_lds) && \
    __has_builtin(__builtin_amdgcn_s_wait_tensorcnt)
#define ATHENA_TDM 1
#else
#define ATHENA_TDM 0
#endif

#define DEV static __device__ __forceinline__

DEV v8f zero8() {
  v8f r;
#pragma unroll
  for (int i = 0; i < 8; ++i) r[i] = 0.0f;
  return r;
}

// A-operand (16x32 bf16, row per lane), contiguous K to match B:
//   lanes 0-15 : row = lane,    K = {0..15}  of the 32-chunk
//   lanes 16-31: row = lane-16, K = {16..31}
// Caller passes base already offset by (hi ? 16 : 0); chunks at +0 and +8.
DEV v16bf load_a16(const __bf16* base) {
  v8bf lo = *(const v8bf*)(base);
  v8bf hi = *(const v8bf*)(base + 8);
  v16bf r;
#pragma unroll
  for (int i = 0; i < 8; ++i) { r[i] = lo[i]; r[8 + i] = hi[i]; }
  return r;
}

// B-operand from LDS (16 contiguous K-values per lane) as two 16B loads so
// only 16-byte alignment is required (padded rows are 16B-aligned).
DEV v16bf load_b16(const __bf16* base) {
  v8bf lo = *(const v8bf*)(base);
  v8bf hi = *(const v8bf*)(base + 8);
  v16bf r;
#pragma unroll
  for (int i = 0; i < 8; ++i) { r[i] = lo[i]; r[8 + i] = hi[i]; }
  return r;
}

DEV v8f wmma_bf16(v16bf a, v16bf b, v8f c) {
  return __builtin_amdgcn_wmma_f32_16x16x32_bf16(
      /*neg_a=*/false, a, /*neg_b=*/false, b,
      /*c_mod=*/(short)0, c, /*reuse_a=*/false, /*reuse_b=*/false);
}

// Row reductions across the 16 lanes holding one C/D row (wave32).
DEV float rowmax16(float v) {
  v = fmaxf(v, __shfl_xor(v, 1, 32));
  v = fmaxf(v, __shfl_xor(v, 2, 32));
  v = fmaxf(v, __shfl_xor(v, 4, 32));
  v = fmaxf(v, __shfl_xor(v, 8, 32));
  return v;
}
DEV float rowsum16(float v) {
  v += __shfl_xor(v, 1, 32);
  v += __shfl_xor(v, 2, 32);
  v += __shfl_xor(v, 4, 32);
  v += __shfl_xor(v, 8, 32);
  return v;
}

// ---------------------------------------------------------------------------
// Tensor DMA descriptor builders (CDNA5 ISA).
// ---------------------------------------------------------------------------
DEV unsigned lds_byte_off(const void* p) {
  // Flat LDS aperture address: bits[31:0] are the LDS byte offset.
  return (unsigned)(uintptr_t)p;
}

DEV v4u tdm_group0(unsigned lds_off, const void* gptr) {
  unsigned long long ga = (unsigned long long)(uintptr_t)gptr;
  v4u g;
  g[0] = 1u;                                         // count=1, user D#
  g[1] = lds_off;                                    // lds_addr
  g[2] = (unsigned)(ga & 0xFFFFFFFFu);               // global_addr[31:0]
  g[3] = (unsigned)((ga >> 32) & 0x01FFFFFFu) | (2u << 30);  // [56:32]|type=2
  return g;
}

// 2D tile, data_size = 8B (code 3), pad_enable = 1.
DEV v8i_ tdm_group1(unsigned tensor_d0, unsigned tensor_d1,
                    unsigned tile_d0, unsigned tile_d1,
                    unsigned long long stride0,
                    unsigned pad_interval, unsigned pad_amount) {
  v8i_ g;
  g[0] = (int)((3u << 16) | (1u << 20) | (pad_interval << 22) |
               (pad_amount << 25));
  g[1] = (int)((tensor_d0 & 0xFFFFu) << 16);
  g[2] = (int)(((tensor_d0 >> 16) & 0xFFFFu) | ((tensor_d1 & 0xFFFFu) << 16));
  g[3] = (int)(((tensor_d1 >> 16) & 0xFFFFu) | ((tile_d0 & 0xFFFFu) << 16));
  g[4] = (int)(tile_d1 & 0xFFFFu);                   // tile_dim1, tile_dim2=0
  g[5] = (int)(stride0 & 0xFFFFFFFFull);
  g[6] = (int)((stride0 >> 32) & 0xFFFFull);         // dim1_stride unused (2D)
  g[7] = 0;
  return g;
}

DEV void tdm_load(v4u g0, v8i_ g1) {
#if ATHENA_TDM
  // 6-arg form (clang-23 / therock-10.0 headers):
  //   (v4u g0, v8i g1, v4i, v4i, v8i, i32 cpol)
  v4i_ z4 = {0, 0, 0, 0};           // groups 2/3 unused for 2D tensors
  v8i_ z8 = {0, 0, 0, 0, 0, 0, 0, 0};
  __builtin_amdgcn_tensor_load_to_lds(g0, g1, z4, z4, z8, 0);
#else
  (void)g0; (void)g1;
#endif
}

template <int N>
DEV void tdm_wait() {
#if ATHENA_TDM
  __builtin_amdgcn_s_wait_tensorcnt(N);  // needs an ICE -> template param
#endif
}

// ---------------------------------------------------------------------------
// Prep: fp32 -> bf16 elementwise convert.
// ---------------------------------------------------------------------------
__global__ void __launch_bounds__(256)
cvt_bf16_kernel(const float* __restrict__ in, __bf16* __restrict__ out, int n) {
  const int i = blockIdx.x * 256 + threadIdx.x;
  if (i < n) out[i] = (__bf16)in[i];
}

// Prep: 1024x1024 fp32 transpose + bf16 convert: out[r][c] = in[c][r].
__global__ void __launch_bounds__(256)
transpose_cvt_kernel(const float* __restrict__ in, __bf16* __restrict__ out) {
  __shared__ float tile[32][33];
  const int tx = threadIdx.x & 31, ty = threadIdx.x >> 5;
  const int r0 = blockIdx.y * 32, c0 = blockIdx.x * 32;
#pragma unroll
  for (int i = 0; i < 32; i += 8)
    tile[ty + i][tx] = in[(size_t)(c0 + ty + i) * 1024 + r0 + tx];
  __syncthreads();
#pragma unroll
  for (int i = 0; i < 32; i += 8)
    out[(size_t)(r0 + ty + i) * 1024 + c0 + tx] = (__bf16)tile[tx][ty + i];
}

// ---------------------------------------------------------------------------
// Fused QKV projection (WMMA) + fractional RoPE.
// grid = (T/16, H, B), block = 256 (8 waves). Wave w owns d-columns
// [16w, 16w+16) of q/k/v. Outputs: qb, kb row-major [b,h,t,d] (post-RoPE);
// vtb transposed [b,h,d,t] -> N-major B operand for P*V.
// ---------------------------------------------------------------------------
__global__ void __launch_bounds__(256)
qkv_rope_kernel(const __bf16* __restrict__ xb,
                const __bf16* __restrict__ wqT, const __bf16* __restrict__ wkT,
                const __bf16* __restrict__ wvT,
                __bf16* __restrict__ qb, __bf16* __restrict__ kb,
                __bf16* __restrict__ vtb) {
  __shared__ __align__(16) float qs[16][128];
  __shared__ __align__(16) float ks[16][128];
  const int wave = threadIdx.x >> 5;
  const int lane = threadIdx.x & 31;
  const int hi = lane >> 4, l15 = lane & 15, rhalf = hi * 8;
  const int b = blockIdx.z, h = blockIdx.y, t0 = blockIdx.x * 16;

  const __bf16* xrow = xb + ((size_t)b * Tc + t0 + l15) * Mc + (hi ? 16 : 0);
  const int nrow = h * Dc + wave * 16 + l15;
  const __bf16* wqr = wqT + (size_t)nrow * Mc + (hi ? 16 : 0);
  const __bf16* wkr = wkT + (size_t)nrow * Mc + (hi ? 16 : 0);
  const __bf16* wvr = wvT + (size_t)nrow * Mc + (hi ? 16 : 0);

  v8f qacc = zero8(), kacc = zero8(), vacc = zero8();
  for (int m0 = 0; m0 < Mc; m0 += 32) {
    const v16bf a = load_a16(xrow + m0);
    qacc = wmma_bf16(a, *(const v16bf*)(wqr + m0), qacc);
    kacc = wmma_bf16(a, *(const v16bf*)(wkr + m0), kacc);
    vacc = wmma_bf16(a, *(const v16bf*)(wvr + m0), vacc);
  }

  // V: store transposed straight from C-layout regs; Q/K: stage in LDS f32.
  const size_t vbase =
      ((size_t)(b * Hc + h) * Dc + wave * 16 + l15) * Tc + t0 + rhalf;
#pragma unroll
  for (int r = 0; r < 8; ++r) {
    qs[r + rhalf][wave * 16 + l15] = qacc[r];
    ks[r + rhalf][wave * 16 + l15] = kacc[r];
    vtb[vbase + r] = (__bf16)vacc[r];
  }
  __syncthreads();

  // Fractional RoPE: rotary over dims [0,64) with half=32, passthrough rest.
  const size_t qkbase = ((size_t)(b * Hc + h) * Tc + t0) * Dc;
  for (int e = threadIdx.x; e < 16 * 128; e += 256) {
    const int row = e >> 7, col = e & 127;
    float qv, kv;
    if (col < 64) {
      const int j = col & 31;
      const float freq = __powf(10000.0f, -(float)j * (1.0f / 32.0f));
      float sn, cs;
      __sincosf((float)(t0 + row) * freq, &sn, &cs);
      if (col < 32) {
        qv = qs[row][col] * cs - qs[row][col + 32] * sn;
        kv = ks[row][col] * cs - ks[row][col + 32] * sn;
      } else {
        qv = qs[row][col - 32] * sn + qs[row][col] * cs;
        kv = ks[row][col - 32] * sn + ks[row][col] * cs;
      }
    } else {
      qv = qs[row][col];
      kv = ks[row][col];
    }
    qb[qkbase + (size_t)row * Dc + col] = (__bf16)qv;
    kb[qkbase + (size_t)row * Dc + col] = (__bf16)kv;
  }
}

// ---------------------------------------------------------------------------
// Flash attention (causal, online softmax). grid = (T/64, B*H), block = 128.
// The block's 4 waves share (b,h); K/V 32-key tiles are DMA'd into LDS by
// wave 0 via tensor_load_to_lds (double-buffered, TENSORcnt pipeline).
// Wave w owns q rows [t0, t0+16); it skips compute for key tiles past its
// causal bound but still participates in the barriers.
// ---------------------------------------------------------------------------
__global__ void __launch_bounds__(128)
attn_kernel(const __bf16* __restrict__ qb, const __bf16* __restrict__ kb,
            const __bf16* __restrict__ vtb, __bf16* __restrict__ obuf) {
  __shared__ __align__(16) __bf16 Kt[2][32][kKRow];   // 32 keys x 128 d (+pad)
  __shared__ __align__(16) __bf16 Vt[2][128][kVRow];  // 128 d x 32 keys (+pad)
  __shared__ __align__(32) __bf16 Pb[4][16 * 32];     // per-wave P transpose
  const int wave = threadIdx.x >> 5;
  const int lane = threadIdx.x & 31;
  const int hi = lane >> 4, l15 = lane & 15, rhalf = hi * 8;
  const int bh = blockIdx.y;
  const int b = bh >> 3, h = bh & 7;
  const int t0 = blockIdx.x * 64 + wave * 16;
  const int niter = blockIdx.x * 2 + 2;  // uniform: keys [0, blk*64+64)

  // Q tile held in registers for the whole key loop (4 A-frags over D=128).
  const __bf16* qrow = qb + ((size_t)bh * Tc + t0 + l15) * Dc + (hi ? 16 : 0);
  v16bf qa[4];
#pragma unroll
  for (int c = 0; c < 4; ++c) qa[c] = load_a16(qrow + c * 32);

  v8f O[8];
#pragma unroll
  for (int nt = 0; nt < 8; ++nt) O[nt] = zero8();
  float mrow[8], lrow[8];
#pragma unroll
  for (int r = 0; r < 8; ++r) { mrow[r] = -3.0e38f; lrow[r] = 0.0f; }

  const __bf16* kbase = kb + (size_t)bh * Tc * Dc;
  const __bf16* vbase = vtb + (size_t)bh * Dc * Tc;

#if ATHENA_TDM
  if (wave == 0) {  // prologue: DMA tile 0 into buffer 0
    tdm_load(tdm_group0(lds_byte_off(&Kt[0][0][0]), kbase),
             tdm_group1(/*td0=*/32, /*td1=*/Tc, /*tile0=*/32, /*tile1=*/32,
                        /*stride0=*/32, /*pad_int=*/5, /*pad_amt=*/3));
    tdm_load(tdm_group0(lds_byte_off(&Vt[0][0][0]), vbase),
             tdm_group1(/*td0=*/512, /*td1=*/Dc, /*tile0=*/8, /*tile1=*/128,
                        /*stride0=*/512, /*pad_int=*/3, /*pad_amt=*/3));
  }
#endif

  for (int it = 0, j0 = 0; it < niter; ++it, j0 += 32) {
    const int cur = it & 1;
#if ATHENA_TDM
    if (wave == 0) {
      if (it + 1 < niter) {  // DMA next tile into the other buffer
        const int nx = cur ^ 1;
        tdm_load(tdm_group0(lds_byte_off(&Kt[nx][0][0]),
                            kbase + (size_t)(j0 + 32) * Dc),
                 tdm_group1(32, Tc, 32, 32, 32, 5, 3));
        tdm_load(tdm_group0(lds_byte_off(&Vt[nx][0][0]), vbase + (j0 + 32)),
                 tdm_group1(512, Dc, 8, 128, 512, 3, 3));
        tdm_wait<2>();  // current tile's 2 DMAs done (in-order TENSORcnt)
      } else {
        tdm_wait<0>();  // last tile: drain everything
      }
    }
    __syncthreads();
#else
    // Fallback: cooperative copy of tile `cur` (same padded LDS layout).
    for (int idx = threadIdx.x; idx < 32 * 8; idx += 128) {
      const int row = idx >> 3, ch = idx & 7;
      *(v8bf*)&Kt[cur][row][ch * 16] =
          *(const v8bf*)(kbase + (size_t)(j0 + row) * Dc + ch * 16);
      *(v8bf*)&Kt[cur][row][ch * 16 + 8] =
          *(const v8bf*)(kbase + (size_t)(j0 + row) * Dc + ch * 16 + 8);
    }
    for (int idx = threadIdx.x; idx < 128 * 4; idx += 128) {
      const int row = idx >> 2, ch = idx & 3;
      *(v8bf*)&Vt[cur][row][ch * 8] =
          *(const v8bf*)(vbase + (size_t)row * Tc + j0 + ch * 8);
    }
    __syncthreads();
#endif

    if (j0 < t0 + 16) {  // wave-uniform causal skip
      // ---- S = Q * K^T from LDS (two 16-key tiles, 4 d-chunks each) ----
      v8f s0 = zero8(), s1 = zero8();
#pragma unroll
      for (int c = 0; c < 4; ++c) {
        s0 = wmma_bf16(qa[c], load_b16(&Kt[cur][l15][c * 32 + hi * 16]), s0);
        s1 = wmma_bf16(qa[c], load_b16(&Kt[cur][16 + l15][c * 32 + hi * 16]),
                       s1);
      }

      // ---- causal mask + online softmax ----
      float alpha[8];
#pragma unroll
      for (int r = 0; r < 8; ++r) {
        const int row = t0 + r + rhalf;
        const float a0 = (j0 + l15      > row) ? kNegInf : s0[r] * kScale;
        const float a1 = (j0 + 16 + l15 > row) ? kNegInf : s1[r] * kScale;
        const float mv = rowmax16(fmaxf(a0, a1));
        const float nm = fmaxf(mrow[r], mv);
        const float al = __expf(mrow[r] - nm);
        const float p0 = __expf(a0 - nm);
        const float p1 = __expf(a1 - nm);
        lrow[r] = lrow[r] * al + rowsum16(p0 + p1);
        mrow[r] = nm;
        alpha[r] = al;
        Pb[wave][(r + rhalf) * 32 + l15]      = (__bf16)p0;
        Pb[wave][(r + rhalf) * 32 + 16 + l15] = (__bf16)p1;
      }
#pragma unroll
      for (int nt = 0; nt < 8; ++nt) {
#pragma unroll
        for (int r = 0; r < 8; ++r) O[nt][r] *= alpha[r];
      }

      // wave-internal LDS RAW: DS in-order; wait, then gather A-layout.
      asm volatile("s_wait_dscnt 0x0" ::: "memory");
      const v16bf pA = load_a16(&Pb[wave][l15 * 32 + (hi ? 16 : 0)]);

      // ---- O += P * V from LDS ----
#pragma unroll
      for (int nt = 0; nt < 8; ++nt)
        O[nt] = wmma_bf16(pA, load_b16(&Vt[cur][nt * 16 + l15][hi * 16]),
                          O[nt]);
    }
    __syncthreads();  // all reads of buf `cur` done before it is re-filled
  }

  // ---- normalize and emit O as [B, T, M] bf16 for the output GEMM ----
#pragma unroll
  for (int r = 0; r < 8; ++r) {
    const float inv = 1.0f / lrow[r];
    const size_t orow =
        ((size_t)b * Tc + t0 + r + rhalf) * Mc + (size_t)h * Dc + l15;
#pragma unroll
    for (int nt = 0; nt < 8; ++nt)
      obuf[orow + nt * 16] = (__bf16)(O[nt][r] * inv);
  }
}

// ---------------------------------------------------------------------------
// Output projection: out[B*T, M] = O[B*T, M] @ wo[M, M] (woT is N-major).
// grid = (B*T/16, M/128), block = 256 (8 waves, one 16-col tile each).
// ---------------------------------------------------------------------------
__global__ void __launch_bounds__(256)
oproj_kernel(const __bf16* __restrict__ obuf, const __bf16* __restrict__ woT,
             float* __restrict__ out) {
  const int wave = threadIdx.x >> 5, lane = threadIdx.x & 31;
  const int hi = lane >> 4, l15 = lane & 15, rhalf = hi * 8;
  const int row0 = blockIdx.x * 16;
  const int col0 = blockIdx.y * 128 + wave * 16;
  const __bf16* arow = obuf + ((size_t)row0 + l15) * Mc + (hi ? 16 : 0);
  const __bf16* brow = woT + (size_t)(col0 + l15) * Mc + (hi ? 16 : 0);
  v8f acc = zero8();
  for (int m0 = 0; m0 < Mc; m0 += 32)
    acc = wmma_bf16(load_a16(arow + m0), *(const v16bf*)(brow + m0), acc);
#pragma unroll
  for (int r = 0; r < 8; ++r)
    out[((size_t)row0 + r + rhalf) * Mc + col0 + l15] = acc[r];
}

// ---------------------------------------------------------------------------
extern "C" void kernel_launch(void* const* d_in, const int* in_sizes, int n_in,
                              void* d_out, int out_size, void* d_ws,
                              size_t ws_size, hipStream_t stream) {
  (void)in_sizes; (void)n_in; (void)out_size; (void)ws_size;
  const float* x  = (const float*)d_in[0];
  const float* wq = (const float*)d_in[1];
  const float* wk = (const float*)d_in[2];
  const float* wv = (const float*)d_in[3];
  const float* wo = (const float*)d_in[4];
  float* out = (float*)d_out;

  char* ws = (char*)d_ws;
  size_t off = 0;
  auto take = [&](size_t elems) -> __bf16* {
    __bf16* p = (__bf16*)(ws + off);
    off += elems * sizeof(__bf16);
    return p;
  };
  const size_t nX = (size_t)Bc * Tc * Mc;      // 4 Mi elems
  const size_t nW = (size_t)Mc * Hc * Dc;      // 1 Mi elems
  const size_t nQ = (size_t)Bc * Hc * Tc * Dc; // 4 Mi elems
  __bf16* xb   = take(nX);
  __bf16* wqT  = take(nW);
  __bf16* wkT  = take(nW);
  __bf16* wvT  = take(nW);
  __bf16* woT  = take(nW);
  __bf16* qb   = take(nQ);
  __bf16* kb   = take(nQ);
  __bf16* vtb  = take(nQ);
  __bf16* obuf = take(nX);  // total ~48 MB of d_ws

  cvt_bf16_kernel<<<(int)((nX + 255) / 256), 256, 0, stream>>>(x, xb, (int)nX);
  const dim3 tgrid(32, 32);
  transpose_cvt_kernel<<<tgrid, 256, 0, stream>>>(wq, wqT);  // [M,HD]->[HD,M]
  transpose_cvt_kernel<<<tgrid, 256, 0, stream>>>(wk, wkT);
  transpose_cvt_kernel<<<tgrid, 256, 0, stream>>>(wv, wvT);
  transpose_cvt_kernel<<<tgrid, 256, 0, stream>>>(wo, woT);  // [HD,M]->[M,HD]

  qkv_rope_kernel<<<dim3(Tc / 16, Hc, Bc), 256, 0, stream>>>(
      xb, wqT, wkT, wvT, qb, kb, vtb);
  attn_kernel<<<dim3(Tc / 64, Bc * Hc), 128, 0, stream>>>(qb, kb, vtb, obuf);
  oproj_kernel<<<dim3(Bc * Tc / 16, Mc / 128), 256, 0, stream>>>(obuf, woT,
                                                                 out);
}
